// QuantumInterferenceAttention_38826504356527
// MI455X (gfx1250) — matrix operand order repaired and checked by reading, not compiled, not run on hardware
//
#include <hip/hip_runtime.h>
#include <math.h>

#define LSEQ 2048
#define DIMM 512
#define NH   8
#define HD   64
#define QKD  1024   // 2*DIM
#define D2   128    // decomposed head dim (cos||sin)

typedef _Float16 h8   __attribute__((ext_vector_type(8)));
typedef _Float16 h16  __attribute__((ext_vector_type(16)));
typedef float    f8   __attribute__((ext_vector_type(8)));
typedef unsigned int u32x4 __attribute__((ext_vector_type(4)));
typedef int          i32x8 __attribute__((ext_vector_type(8)));
typedef int          i32x4 __attribute__((ext_vector_type(4)));

static __device__ __forceinline__ h16 cat8(h8 lo, h8 hi) {
  return __builtin_shufflevector(lo, hi, 0,1,2,3,4,5,6,7,8,9,10,11,12,13,14,15);
}

// ---------------- prep: f16 conversions + weight transposes ----------------
__global__ __launch_bounds__(256)
void prep_kernel(const float* __restrict__ x, const float* __restrict__ Wqs,
                 const float* __restrict__ Wout, _Float16* __restrict__ xh,
                 _Float16* __restrict__ WqsT, _Float16* __restrict__ WoutT) {
  int idx = blockIdx.x * blockDim.x + threadIdx.x;
  if (idx < LSEQ * DIMM) xh[idx] = (_Float16)x[idx];
  if (idx < QKD * DIMM) {                 // WqsT[n][k] = Wqs[k][n]
    int n = idx >> 9, k = idx & 511;
    WqsT[idx] = (_Float16)Wqs[k * QKD + n];
  }
  if (idx < DIMM * DIMM) {                // WoutT[n][k] = Wout[k][n]
    int n = idx >> 9, k = idx & 511;
    WoutT[idx] = (_Float16)Wout[k * DIMM + n];
  }
}

// ---------------- GEMM1: qs = x @ W_qs + b_qs  (2048x1024, K=512) ----------
__global__ __launch_bounds__(128)
void gemm_qs_kernel(const _Float16* __restrict__ xh, const _Float16* __restrict__ WqsT,
                    const float* __restrict__ bqs, float* __restrict__ qs) {
  const int lane = threadIdx.x & 31, wave = threadIdx.x >> 5;
  const int rowbase = blockIdx.x * 16;
  const int colbase = blockIdx.y * 256 + wave * 64;
  const int m = lane & 15, half = lane >> 4;

  f8 acc[4] = {};
  const _Float16* arow = xh + (size_t)(rowbase + m) * DIMM;
  for (int c = 0; c < 16; ++c) {
    h8 alo = *(const h8*)(arow + c * 32 + half * 16);
    h8 ahi = *(const h8*)(arow + c * 32 + half * 16 + 8);
    h16 a = cat8(alo, ahi);
    #pragma unroll
    for (int t = 0; t < 4; ++t) {
      h16 b = *(const h16*)(WqsT + (size_t)(colbase + t * 16 + m) * DIMM + c * 32 + half * 16);
      acc[t] = __builtin_amdgcn_wmma_f32_16x16x32_f16(false, a, false, b, (short)0, acc[t], false, false);
    }
  }
  #pragma unroll
  for (int t = 0; t < 4; ++t) {
    int col = colbase + t * 16 + m;
    float bias = bqs[col];
    #pragma unroll
    for (int i = 0; i < 8; ++i)
      qs[(size_t)(rowbase + half * 8 + i) * QKD + col] = acc[t][i] + bias;
  }
}

// ---------------- LN + sigmoid/tanh + trig decomposition pack --------------
__global__ __launch_bounds__(256)
void lnpack_kernel(const float* __restrict__ qs, const float* __restrict__ lnw,
                   const float* __restrict__ lnb, const float* __restrict__ freq,
                   const float* __restrict__ ps, const float* __restrict__ x,
                   _Float16* __restrict__ Qp, _Float16* __restrict__ Kp,
                   _Float16* __restrict__ Vt) {
  __shared__ float red[256], red2[256], amp_s[512], ph_s[512];
  const int row = blockIdx.x, tid = threadIdx.x;
  const float* q = qs + (size_t)row * QKD;

  float v[4], s0 = 0.f, s1 = 0.f;
  #pragma unroll
  for (int j = 0; j < 4; ++j) { float t = q[tid + j * 256]; v[j] = t; s0 += t; s1 += t * t; }
  red[tid] = s0; red2[tid] = s1; __syncthreads();
  for (int s = 128; s > 0; s >>= 1) {
    if (tid < s) { red[tid] += red[tid + s]; red2[tid] += red2[tid + s]; }
    __syncthreads();
  }
  float mean = red[0] * (1.f / 1024.f);
  float var  = red2[0] * (1.f / 1024.f) - mean * mean;
  float rstd = rsqrtf(var + 1e-5f);
  __syncthreads();

  #pragma unroll
  for (int j = 0; j < 4; ++j) {
    int c = tid + j * 256;
    float t = (v[j] - mean) * rstd * lnw[c] + lnb[c];
    if (c < 512) amp_s[c] = 1.f / (1.f + __expf(-t));
    else         ph_s[c - 512] = tanhf(t) * 3.14159265358979323846f;
  }
  __syncthreads();
  float a0 = amp_s[tid], a1 = amp_s[tid + 256];
  red[tid] = a0 * a0 + a1 * a1; __syncthreads();
  for (int s = 128; s > 0; s >>= 1) {
    if (tid < s) red[tid] += red[tid + s];
    __syncthreads();
  }
  float inorm = 1.f / (sqrtf(red[0]) + 1e-8f);

  #pragma unroll
  for (int j = 0; j < 2; ++j) {
    int d = tid + j * 256;
    int h = d >> 6, dl = d & 63;
    float a = amp_s[d] * inorm, p = ph_s[d];
    float f = freq[h], sh = ps[h];
    float sq, cq, sk, ck;
    __sincosf(p * f + sh, &sq, &cq);
    __sincosf(p * f, &sk, &ck);
    size_t base = ((size_t)h * LSEQ + row) * D2;
    Qp[base + dl]      = (_Float16)(a * cq);
    Qp[base + 64 + dl] = (_Float16)(a * sq);
    Kp[base + dl]      = (_Float16)(a * ck);
    Kp[base + 64 + dl] = (_Float16)(a * sk);
    Vt[((size_t)h * HD + dl) * LSEQ + row] = (_Float16)x[(size_t)row * DIMM + d];
  }
}

// ---------------- fused attention: S=QpKp^T, softmax (no max), O=P V ------
// K blocks are staged into LDS once per workgroup by the Tensor Data Mover
// (TENSOR_LOAD_TO_LDS, tracked with TENSORcnt), then consumed by 4 waves.
__global__ __launch_bounds__(128)
void attn_kernel(const _Float16* __restrict__ Qp, const _Float16* __restrict__ Kp,
                 const _Float16* __restrict__ Vt, float* __restrict__ AO) {
  __shared__ _Float16 klds[32][D2];       // 8KB K block (TDM destination)
  __shared__ _Float16 plds[4][16][32];    // per-wave C->A transpose staging
  const int lane = threadIdx.x & 31, wave = threadIdx.x >> 5;
  const int h = blockIdx.y;
  const int qbase = blockIdx.x * 64 + wave * 16;
  const int m = lane & 15, half = lane >> 4;

  const _Float16* qrow = Qp + ((size_t)h * LSEQ + qbase + m) * D2;
  h16 aq[4];
  #pragma unroll
  for (int c = 0; c < 4; ++c) {
    h8 lo = *(const h8*)(qrow + c * 32 + half * 16);
    h8 hi = *(const h8*)(qrow + c * 32 + half * 16 + 8);
    aq[c] = cat8(lo, hi);
  }
  h16 ones;
  #pragma unroll
  for (int j = 0; j < 16; ++j) ones[j] = (_Float16)1.0f;

  f8 o[4] = {};
  f8 ls = {};
  const _Float16* kbaseP = Kp + (size_t)h * LSEQ * D2;
  const _Float16* vbaseP = Vt + (size_t)h * HD * LSEQ;

  // Tensor DMA descriptor group 1 (constant): data_size=2B, 2D tile 128x32,
  // tensor_dim0=128, tensor_dim1=2048, dim0 stride=128 elements.
  const i32x8 g1 = { (int)0x00010000,          // data_size=1 (2B), mask=0
                     (int)(128u << 16),        // tensor_dim0[15:0]
                     (int)(2048u << 16),       // tensor_dim0[31:16] | tensor_dim1[15:0]
                     (int)(128u << 16),        // tensor_dim1[31:16] | tile_dim0
                     32,                       // tile_dim1=32 keys, tile_dim2=0
                     128,                      // tensor_dim0_stride[31:0]
                     0, 0 };
  const i32x4 z4 = { 0, 0, 0, 0 };
  const i32x8 z8 = { 0, 0, 0, 0, 0, 0, 0, 0 };
  const unsigned klds_off = (unsigned)(uintptr_t)&klds[0][0]; // flat LDS addr[31:0]

  for (int kb = 0; kb < LSEQ / 32; ++kb) {
    const int key0 = kb * 32;
    if (wave == 0) {
      unsigned long long ga = (unsigned long long)(uintptr_t)(kbaseP + (size_t)key0 * D2);
      u32x4 g0;
      g0.x = 1u;                                           // count=1, user mode
      g0.y = klds_off;                                     // lds_addr
      g0.z = (unsigned)ga;                                 // global_addr[31:0]
      g0.w = ((unsigned)(ga >> 32) & 0x01FFFFFFu) | (2u << 30); // addr[56:32] | type=2
      __builtin_amdgcn_tensor_load_to_lds(g0, g1, z4, z4, z8, 0);
      __builtin_amdgcn_s_wait_tensorcnt(0);
    }
    if (kb + 1 < LSEQ / 32)
      __builtin_prefetch(vbaseP + (size_t)m * LSEQ + key0 + 32, 0, 1);
    __syncthreads();                      // K block visible to all waves
    #pragma unroll
    for (int t = 0; t < 2; ++t) {
      f8 s = {};
      const _Float16* kr = &klds[t * 16 + m][half * 16];
      #pragma unroll
      for (int c = 0; c < 4; ++c) {
        h16 b = *(const h16*)(kr + c * 32);
        s = __builtin_amdgcn_wmma_f32_16x16x32_f16(false, aq[c], false, b, (short)0, s, false, false);
      }
      // |s|<=1 so exp(s/8) can never overflow: no max-subtraction needed
      #pragma unroll
      for (int i = 0; i < 8; ++i)
        plds[wave][half * 8 + i][t * 16 + m] = (_Float16)__expf(s[i] * 0.125f);
    }
    __syncthreads();                      // P staged; klds fully consumed
    h8 plo = *(const h8*)&plds[wave][m][half * 16];
    h8 phi = *(const h8*)&plds[wave][m][half * 16 + 8];
    h16 pa = cat8(plo, phi);
    ls = __builtin_amdgcn_wmma_f32_16x16x32_f16(false, pa, false, ones, (short)0, ls, false, false);
    #pragma unroll
    for (int nt = 0; nt < 4; ++nt) {
      h16 bv = *(const h16*)(vbaseP + (size_t)(nt * 16 + m) * LSEQ + key0 + half * 16);
      o[nt] = __builtin_amdgcn_wmma_f32_16x16x32_f16(false, pa, false, bv, (short)0, o[nt], false, false);
    }
  }
  #pragma unroll
  for (int nt = 0; nt < 4; ++nt)
    #pragma unroll
    for (int i = 0; i < 8; ++i)
      AO[(size_t)(qbase + half * 8 + i) * DIMM + h * HD + nt * 16 + m] = o[nt][i] / ls[i];
}

// ---------------- GEMM2: out = AO @ W_out + b_out (2048x512, K=512) -------
__global__ __launch_bounds__(128)
void gemm_out_kernel(const float* __restrict__ AO, const _Float16* __restrict__ WoutT,
                     const float* __restrict__ bout, float* __restrict__ out) {
  const int lane = threadIdx.x & 31, wave = threadIdx.x >> 5;
  const int rowbase = blockIdx.x * 16;
  const int colbase = blockIdx.y * 256 + wave * 64;
  const int m = lane & 15, half = lane >> 4;

  f8 acc[4] = {};
  const float* arow = AO + (size_t)(rowbase + m) * DIMM;
  for (int c = 0; c < 16; ++c) {
    f8 alo = *(const f8*)(arow + c * 32 + half * 16);
    f8 ahi = *(const f8*)(arow + c * 32 + half * 16 + 8);
    h16 a = cat8(__builtin_convertvector(alo, h8), __builtin_convertvector(ahi, h8));
    #pragma unroll
    for (int t = 0; t < 4; ++t) {
      h16 b = *(const h16*)(WoutT + (size_t)(colbase + t * 16 + m) * DIMM + c * 32 + half * 16);
      acc[t] = __builtin_amdgcn_wmma_f32_16x16x32_f16(false, a, false, b, (short)0, acc[t], false, false);
    }
  }
  #pragma unroll
  for (int t = 0; t < 4; ++t) {
    int col = colbase + t * 16 + m;
    float bias = bout[col];
    #pragma unroll
    for (int i = 0; i < 8; ++i)
      out[(size_t)(rowbase + half * 8 + i) * DIMM + col] = acc[t][i] + bias;
  }
}

extern "C" void kernel_launch(void* const* d_in, const int* in_sizes, int n_in,
                              void* d_out, int out_size, void* d_ws, size_t ws_size,
                              hipStream_t stream) {
  (void)in_sizes; (void)n_in; (void)out_size; (void)ws_size;
  const float* x    = (const float*)d_in[0];
  const float* Wqs  = (const float*)d_in[1];
  const float* bqs  = (const float*)d_in[2];
  const float* lnw  = (const float*)d_in[3];
  const float* lnb  = (const float*)d_in[4];
  const float* freq = (const float*)d_in[5];
  const float* ps   = (const float*)d_in[6];
  const float* Wout = (const float*)d_in[7];
  const float* bout = (const float*)d_in[8];
  // d_in[9] (cos_table) is analytically replaced by the cos/sin decomposition.

  char* ws = (char*)d_ws;
  size_t off = 0;
  float*    qs    = (float*)(ws + off);    off += (size_t)LSEQ * QKD * 4;     // 8 MB
  _Float16* xh    = (_Float16*)(ws + off); off += (size_t)LSEQ * DIMM * 2;    // 2 MB
  _Float16* WqsT  = (_Float16*)(ws + off); off += (size_t)QKD * DIMM * 2;     // 1 MB
  _Float16* WoutT = (_Float16*)(ws + off); off += (size_t)DIMM * DIMM * 2;    // 0.5 MB
  _Float16* Qp    = (_Float16*)(ws + off); off += (size_t)NH * LSEQ * D2 * 2; // 4 MB
  _Float16* Kp    = (_Float16*)(ws + off); off += (size_t)NH * LSEQ * D2 * 2; // 4 MB
  _Float16* Vt    = (_Float16*)(ws + off); off += (size_t)NH * HD * LSEQ * 2; // 2 MB
  float*    AO    = (float*)(ws + off);    off += (size_t)LSEQ * DIMM * 4;    // 4 MB

  prep_kernel    <<<4096, 256, 0, stream>>>(x, Wqs, Wout, xh, WqsT, WoutT);
  gemm_qs_kernel <<<dim3(128, 4), 128, 0, stream>>>(xh, WqsT, bqs, qs);
  lnpack_kernel  <<<2048, 256, 0, stream>>>(qs, lnw, lnb, freq, ps, x, Qp, Kp, Vt);
  attn_kernel    <<<dim3(32, 8), 128, 0, stream>>>(Qp, Kp, Vt, AO);
  gemm_out_kernel<<<dim3(128, 2), 128, 0, stream>>>(AO, WoutT, bout, (float*)d_out);
}
